// DeformableStrip_43800076485365
// MI455X (gfx1250) — matrix operand order repaired and checked by reading, not compiled, not run on hardware
//
#include <hip/hip_runtime.h>
#include <cstddef>

typedef __attribute__((ext_vector_type(16))) _Float16 v16h;
typedef __attribute__((ext_vector_type(8)))  float    v8f;

#define HW   65536      // 256*256
#define NB   2
#define CI   64
#define CO   64
#define NP   9          // K*K taps

// ---------------- workspace layout (bytes, 256-aligned) ----------------
static const size_t OFF_XN    = 0;                       // x NHWC f32: 2*65536*64*4 = 33554432
static const size_t OFF_HN    = 33554432;                // h NHWC f32: 2*65536*16*4 = 8388608
static const size_t OFF_OFF   = 41943040;                // offsets NCHW f32: 2*18*65536*4 = 9437184
static const size_t OFF_WZ    = 51380224;                // swizzled weight f16: 4*18*32*16*2 = 73728
static const size_t OFF_OUTP  = 51453952;                // pre-BN out NCHW f32: 33554432
static const size_t OFF_PART  = 85008384;                // partials: 8192*256*4 = 8388608
static const size_t OFF_STATS = 93396992;                // mean[64] + rstd[64]

// ---------------- 1) NCHW -> NHWC transpose of x ----------------
__global__ void k_prep_nhwc(const float* __restrict__ x, float* __restrict__ xn) {
    int e  = blockIdx.x * 256 + threadIdx.x;   // NB*HW*16 threads
    int c4 = e & 15;
    int s  = (e >> 4) & (HW - 1);
    int b  = e >> 20;
    int c0 = c4 * 4;
    const float* xp = x + (size_t)(b * CI + c0) * HW + s;
    float4 v;
    v.x = xp[0]; v.y = xp[HW]; v.z = xp[2 * HW]; v.w = xp[3 * HW];
    *(float4*)(xn + ((size_t)(b * HW + s) * CI + c0)) = v;
}

// ---------------- 2) pre-swizzle weight into WMMA B-fragment order ----------------
// wz[(ntile*18 + kc)*32 + lane] is the 16-half v16h for that K-chunk.
// B(32x16 f16) layout: lane n<16 element e -> (K=(e&7)+16*(e>>3), col=n);
// lane n+16 element e -> (K=8+(e&7)+16*(e>>3), col=n).  kc = p*2 + chalf.
__global__ void k_wswz(const float* __restrict__ w, _Float16* __restrict__ wz) {
    int t = blockIdx.x * 256 + threadIdx.x;
    if (t >= 4 * 18 * 32) return;
    int lane = t & 31;
    int kc   = (t >> 5) % 18;
    int nt   = t / (18 * 32);
    int col  = lane & 15;
    int khi  = lane >> 4;
    int p    = kc >> 1;
    int chb  = (kc & 1) * 32;
    int o    = nt * 16 + col;
    union { v16h v; _Float16 h[16]; } u;
    for (int e = 0; e < 16; ++e) {
        int c = chb + khi * 8 + (e & 7) + ((e >> 3) << 4);
        u.h[e] = (_Float16)w[(o * CI + c) * NP + p];
    }
    *(v16h*)(wz + (size_t)t * 16) = u.v;
}

// ---------------- 3) conv1: 64->16, 3x3, ReLU, NHWC out ----------------
__global__ void k_conv1(const float* __restrict__ xn, const float* __restrict__ w1,
                        float* __restrict__ hn) {
    __shared__ float wl[CI * NP];
    int oc  = blockIdx.y;
    int tid = threadIdx.x;
    for (int i = tid; i < CI * NP; i += 256) wl[i] = w1[oc * CI * NP + i];
    __syncthreads();
    int s  = blockIdx.x * 256 + tid;
    int b  = s >> 16;
    int sp = s & (HW - 1);
    int y = sp >> 8, x = sp & 255;
    float acc = 0.f;
    for (int ky = -1; ky <= 1; ++ky) {
        int yy = y + ky;
        if ((unsigned)yy >= 256u) continue;
        for (int kx = -1; kx <= 1; ++kx) {
            int xx = x + kx;
            if ((unsigned)xx >= 256u) continue;
            int tap = (ky + 1) * 3 + (kx + 1);
            const float4* xp = (const float4*)(xn + ((size_t)(b * HW) + yy * 256 + xx) * CI);
            for (int c0 = 0; c0 < CI; c0 += 4) {
                float4 v = xp[c0 >> 2];
                acc += v.x * wl[(c0 + 0) * NP + tap];
                acc += v.y * wl[(c0 + 1) * NP + tap];
                acc += v.z * wl[(c0 + 2) * NP + tap];
                acc += v.w * wl[(c0 + 3) * NP + tap];
            }
        }
    }
    hn[(size_t)s * 16 + oc] = fmaxf(acc, 0.f);
}

// ---------------- 4) conv2: 16->18, 3x3, offsets NCHW ----------------
__global__ void k_conv2(const float* __restrict__ hn, const float* __restrict__ w2,
                        float* __restrict__ off) {
    __shared__ float wl[16 * NP];
    int oc  = blockIdx.y;
    int tid = threadIdx.x;
    for (int i = tid; i < 16 * NP; i += 256) wl[i] = w2[oc * 16 * NP + i];
    __syncthreads();
    int s  = blockIdx.x * 256 + tid;
    int b  = s >> 16;
    int sp = s & (HW - 1);
    int y = sp >> 8, x = sp & 255;
    float acc = 0.f;
    for (int ky = -1; ky <= 1; ++ky) {
        int yy = y + ky;
        if ((unsigned)yy >= 256u) continue;
        for (int kx = -1; kx <= 1; ++kx) {
            int xx = x + kx;
            if ((unsigned)xx >= 256u) continue;
            int tap = (ky + 1) * 3 + (kx + 1);
            const float4* hp = (const float4*)(hn + ((size_t)(b * HW) + yy * 256 + xx) * 16);
            for (int c0 = 0; c0 < 16; c0 += 4) {
                float4 v = hp[c0 >> 2];
                acc += v.x * wl[(c0 + 0) * NP + tap];
                acc += v.y * wl[(c0 + 1) * NP + tap];
                acc += v.z * wl[(c0 + 2) * NP + tap];
                acc += v.w * wl[(c0 + 3) * NP + tap];
            }
        }
    }
    off[((size_t)(b * 18 + oc) << 16) + sp] = acc;
}

// ---------------- 5) fused deformable gather + WMMA GEMM ----------------
// Branchless bilinear gather of this thread's 8 channels for tap p -> packed f16.
__device__ __forceinline__ float4 gather8(const float* __restrict__ xn,
                                          const float* __restrict__ off,
                                          int b, int sp, int y, int x,
                                          int p, int c0) {
    int ky = p / 3 - 1, kx = p % 3 - 1;
    float oy = off[(((size_t)b * 18 + 2 * p) << 16) + sp];
    float ox = off[(((size_t)b * 18 + 2 * p + 1) << 16) + sp];
    float ys = (float)(y + ky) + oy;
    float xs = (float)(x + kx) + ox;
    float y0f = floorf(ys), x0f = floorf(xs);
    int   y0 = (int)y0f, x0 = (int)x0f;
    float fy = ys - y0f, fx = xs - x0f;
    float a[8] = {0, 0, 0, 0, 0, 0, 0, 0};
#pragma unroll
    for (int t = 0; t < 4; ++t) {
        int dy = t >> 1, dx = t & 1;
        int yi = y0 + dy, xi = x0 + dx;
        float wgt = (dy ? fy : (1.f - fy)) * (dx ? fx : (1.f - fx));
        bool valid = ((unsigned)yi < 256u) && ((unsigned)xi < 256u);
        wgt = valid ? wgt : 0.f;                       // zero weight outside
        int yc = min(max(yi, 0), 255);                 // clamp -> always load
        int xc = min(max(xi, 0), 255);
        const float4* src =
            (const float4*)(xn + (((size_t)(b * HW) + yc * 256 + xc) * CI + c0));
        float4 v0 = src[0], v1 = src[1];
        a[0] += wgt * v0.x; a[1] += wgt * v0.y;
        a[2] += wgt * v0.z; a[3] += wgt * v0.w;
        a[4] += wgt * v1.x; a[5] += wgt * v1.y;
        a[6] += wgt * v1.z; a[7] += wgt * v1.w;
    }
    union { float4 f4; _Float16 h[8]; } pk;
#pragma unroll
    for (int i = 0; i < 8; ++i) pk.h[i] = (_Float16)a[i];
    return pk.f4;
}

// Block = 128 threads (4 waves).  Block owns 16 consecutive spatial positions;
// wave w owns out-channels [16w,16w+16).  K = 9*64 = 576 consumed as 18 chunks
// of 32 via v_wmma_f32_16x16x32_f16.  Ping-pong LDS A buffer: one barrier per
// tap; gather(p+1) overlaps WMMA(p).
__global__ void k_dcn_gemm(const float* __restrict__ xn, const float* __restrict__ off,
                           const _Float16* __restrict__ wz, const float* __restrict__ bias,
                           float* __restrict__ outp, float* __restrict__ partial) {
    __shared__ float4 ldsA[2][16 * 8];   // 2 x (16 rows x 64 f16)

    int tid = threadIdx.x;
    int T   = blockIdx.x;
    int b   = T >> 12;                // 4096 tiles per image
    int s0  = (T & 4095) << 4;        // in-image spatial base
    int sl  = tid >> 3;               // 0..15 row (gather stage)
    int cb  = tid & 7;                // 8-channel block (gather stage)
    int c0  = cb << 3;
    int sp  = s0 + sl;
    int y = sp >> 8, x = sp & 255;

    int lane = tid & 31;
    int wv   = tid >> 5;              // wave == N-tile
    int row  = lane & 15;
    int hi   = lane >> 4;

    v8f c = {0.f, 0.f, 0.f, 0.f, 0.f, 0.f, 0.f, 0.f};

    // prologue: stage tap 0
    ldsA[0][sl * 8 + cb] = gather8(xn, off, b, sp, y, x, 0, c0);

#pragma unroll
    for (int p = 0; p < NP; ++p) {
        __syncthreads();              // buf[p&1] visible; buf[(p+1)&1] free
        if (p + 1 < NP)
            ldsA[(p + 1) & 1][sl * 8 + cb] = gather8(xn, off, b, sp, y, x, p + 1, c0);

        const float4* abuf = ldsA[p & 1];
#pragma unroll
        for (int hf = 0; hf < 2; ++hf) {
            // 16-bit A 16x32 layout: lane<16 holds K {0..7,16..23}; lane>=16 holds
            // K {8..15,24..31} -> two b128 LDS reads 32B apart.
            union { v16h v; float4 f4[2]; } af;
            int pb = row * 8 + hf * 4 + hi;
            af.f4[0] = abuf[pb];
            af.f4[1] = abuf[pb + 2];
            int kc = p * 2 + hf;
            v16h bm = *(const v16h*)(wz + ((size_t)(wv * 18 + kc) * 32 + lane) * 16);
            c = __builtin_amdgcn_wmma_f32_16x16x32_f16(false, af.v, false, bm,
                                                       (short)0, c, false, false);
        }
    }

    // ---- epilogue: bias, pre-BN store, per-channel partial sums ----
    int   o  = wv * 16 + (lane & 15);
    float bo = bias[o];
    float vals[8];
    float s1 = 0.f, s2 = 0.f;
#pragma unroll
    for (int j = 0; j < 8; ++j) {
        float v = c[j] + bo;
        vals[j] = v; s1 += v; s2 += v * v;
    }
    // C layout: VGPR j -> row j (lanes 0-15) / row 8+j (lanes 16-31)
    size_t obase = (((size_t)(b * CO + o)) << 16) + s0 + (hi ? 8 : 0);
    float4* op = (float4*)(outp + obase);
    op[0] = make_float4(vals[0], vals[1], vals[2], vals[3]);
    op[1] = make_float4(vals[4], vals[5], vals[6], vals[7]);
    int slot = o * 2 + hi;
    partial[(size_t)T * 256 + slot]       = s1;
    partial[(size_t)T * 256 + 128 + slot] = s2;
}

// ---------------- 6) reduce partials -> mean/rstd ----------------
__global__ void k_reduce(const float* __restrict__ partial, float* __restrict__ stats) {
    __shared__ float tot[256];
    int tid = threadIdx.x;
    float t = 0.f;
    for (int r = 0; r < 8192; ++r) t += partial[(size_t)r * 256 + tid];
    tot[tid] = t;
    __syncthreads();
    if (tid < 64) {
        float sum = tot[tid * 2] + tot[tid * 2 + 1];
        float sq  = tot[128 + tid * 2] + tot[128 + tid * 2 + 1];
        const float invN = 1.f / 131072.f;
        float mean = sum * invN;
        float var  = sq * invN - mean * mean;
        stats[tid]      = mean;
        stats[64 + tid] = rsqrtf(var + 1e-5f);
    }
}

// ---------------- 7) batchnorm + affine + ReLU ----------------
__global__ void k_bn(const float* __restrict__ outp, const float* __restrict__ stats,
                     const float* __restrict__ gamma, const float* __restrict__ beta,
                     float* __restrict__ out) {
    int idx = blockIdx.x * 256 + threadIdx.x;  // float4 index
    int e   = idx << 2;
    int o   = (e >> 16) & 63;
    float mean = stats[o], rstd = stats[64 + o];
    float scale = rstd * gamma[o];
    float shift = beta[o] - mean * scale;
    float4 v = ((const float4*)outp)[idx];
    v.x = fmaxf(v.x * scale + shift, 0.f);
    v.y = fmaxf(v.y * scale + shift, 0.f);
    v.z = fmaxf(v.z * scale + shift, 0.f);
    v.w = fmaxf(v.w * scale + shift, 0.f);
    ((float4*)out)[idx] = v;
}

extern "C" void kernel_launch(void* const* d_in, const int* in_sizes, int n_in,
                              void* d_out, int out_size, void* d_ws, size_t ws_size,
                              hipStream_t stream) {
    (void)in_sizes; (void)n_in; (void)out_size; (void)ws_size;
    const float* x     = (const float*)d_in[0];
    const float* w1    = (const float*)d_in[1];
    const float* w2    = (const float*)d_in[2];
    const float* w     = (const float*)d_in[3];
    const float* bias  = (const float*)d_in[4];
    const float* gamma = (const float*)d_in[5];
    const float* beta  = (const float*)d_in[6];

    char* ws = (char*)d_ws;
    float*    xn    = (float*)(ws + OFF_XN);
    float*    hn    = (float*)(ws + OFF_HN);
    float*    off   = (float*)(ws + OFF_OFF);
    _Float16* wz    = (_Float16*)(ws + OFF_WZ);
    float*    outp  = (float*)(ws + OFF_OUTP);
    float*    part  = (float*)(ws + OFF_PART);
    float*    stats = (float*)(ws + OFF_STATS);

    k_prep_nhwc<<<8192, 256, 0, stream>>>(x, xn);
    k_wswz     <<<9, 256, 0, stream>>>(w, wz);
    k_conv1    <<<dim3(512, 16), 256, 0, stream>>>(xn, w1, hn);
    k_conv2    <<<dim3(512, 18), 256, 0, stream>>>(hn, w2, off);
    k_dcn_gemm <<<8192, 128, 0, stream>>>(xn, off, wz, bias, outp, part);
    k_reduce   <<<1, 256, 0, stream>>>(part, stats);
    k_bn       <<<8192, 256, 0, stream>>>(outp, stats, gamma, beta, (float*)d_out);
}
